// CNN_56702158241937
// MI455X (gfx1250) — matrix operand order repaired and checked by reading, not compiled, not run on hardware
//
#include <hip/hip_runtime.h>
#include <hip/hip_bf16.h>

typedef _Float16 f16;
typedef _Float16 v8h  __attribute__((ext_vector_type(8)));
typedef _Float16 v16h __attribute__((ext_vector_type(16)));
typedef float    v8f  __attribute__((ext_vector_type(8)));
typedef int      v4i  __attribute__((ext_vector_type(4)));

// Probe the gfx1250 async global->LDS builtins; fall back to plain LDS stores.
#ifndef __has_builtin
#define __has_builtin(x) 0
#endif
#if defined(__HIP_DEVICE_COMPILE__) && \
    __has_builtin(__builtin_amdgcn_global_load_async_to_lds_b128) && \
    __has_builtin(__builtin_amdgcn_s_wait_asynccnt)
#define USE_ASYNC_LDS 1
#else
#define USE_ASYNC_LDS 0
#endif

// Concatenate two 8-halve loads into the 16-halve WMMA operand register set.
static __device__ __forceinline__ v16h ld16h(const f16* p0, const f16* p1) {
    v8h lo = *(const v8h*)p0;
    v8h hi = *(const v8h*)p1;
    return __builtin_shufflevector(lo, hi, 0,1,2,3,4,5,6,7,8,9,10,11,12,13,14,15);
}

// 16-byte global -> LDS copy: async DMA path if available, else through VGPRs.
// Builtin signature (probe-confirmed): (v4i AS1*, v4i AS3*, imm offset, imm cpol).
static __device__ __forceinline__ void copy16(const f16* g, f16* l) {
#if USE_ASYNC_LDS
    __builtin_amdgcn_global_load_async_to_lds_b128(
        (__attribute__((address_space(1))) v4i*)(v4i*)(void*)g,
        (__attribute__((address_space(3))) v4i*)(v4i*)(void*)l, 0, 0);
#else
    *(v8h*)l = *(const v8h*)g;
#endif
}
static __device__ __forceinline__ void wait_stage() {
#if USE_ASYNC_LDS
    __builtin_amdgcn_s_wait_asynccnt(0);
#endif
    __syncthreads();
}

// ---------------------------------------------------------------------------
// Constants
//   x: (64,16,2175); conv1 k=5 -> 2171; conv2 k=5 -> 2167; conv3 k=120 -> L=2048
//   h2f (f16, padded): [n][ci][TP], TP = 2176 (valid t < 2167; tail zeroed)
//   conv3 K-layout: Kidx = ci*128 + k, k in [0,128), k>=120 zero-padded in B
// ---------------------------------------------------------------------------
#define TP 2176
#define BSTR 40  // LDS column stride (halves) for staged B tile: bank-conflict-free

// ---------------------------------------------------------------------------
// Kernel 0: pack weights into WMMA-friendly f16 layouts + zero h2f padding.
// ---------------------------------------------------------------------------
__global__ void prep_kernel(const float* __restrict__ c3w,
                            const float* __restrict__ l1w,
                            const float* __restrict__ outw,
                            f16* __restrict__ Bw, f16* __restrict__ l1f,
                            f16* __restrict__ outf, f16* __restrict__ h2f)
{
    int idx = blockIdx.x * blockDim.x + threadIdx.x;
    if (idx < 128 * 2048) {
        int co = idx >> 11;
        int K  = idx & 2047;
        int ci = K >> 7, k = K & 127;
        float v = (k < 120) ? c3w[((size_t)co * 16 + ci) * 120 + k] : 0.f;
        Bw[(size_t)co * 2048 + K] = (f16)v;
    }
    if (idx < 64 * 128) l1f[idx] = (f16)l1w[idx];
    if (idx < 16 * 64) {
        int o = idx >> 6, k = idx & 63;
        outf[idx] = (o < 5) ? (f16)outw[o * 64 + k] : (f16)0.f;
    }
    if (idx < 1024 * 16) {             // zero h2f tail: t in [2167, TP)
        int row = idx >> 4, j = idx & 15;
        if (j < TP - 2167) h2f[(size_t)row * TP + 2167 + j] = (f16)0.f;
    }
}

// ---------------------------------------------------------------------------
// Kernel 1: fused conv1+conv2 (both 16ch, k=5), ReLU, emit h2 as f16.
// ---------------------------------------------------------------------------
__global__ __launch_bounds__(256) void conv12_kernel(
    const float* __restrict__ x,
    const float* __restrict__ w1, const float* __restrict__ b1,
    const float* __restrict__ w2, const float* __restrict__ b2,
    f16* __restrict__ h2f)
{
    __shared__ __attribute__((aligned(16))) float xs[16 * 264];
    __shared__ __attribute__((aligned(16))) float h1s[16 * 264];
    __shared__ float w1s[1280];
    __shared__ float w2s[1280];

    const int tid   = threadIdx.x;
    const int n     = blockIdx.y;
    const int tBase = blockIdx.x * 256;

    for (int i = tid; i < 1280; i += 256) { w1s[i] = w1[i]; w2s[i] = w2[i]; }
    for (int i = tid; i < 16 * 264; i += 256) {
        int ci = i / 264, p = i - ci * 264;
        int g = tBase + p;
        xs[i] = (g < 2175) ? x[((size_t)n * 16 + ci) * 2175 + g] : 0.f;
    }
    __syncthreads();

    for (int i = tid; i < 16 * 260; i += 256) {   // conv1, halo p in [0,260)
        int co = i & 15, p = i >> 4;
        float v = b1[co];
        #pragma unroll
        for (int ci = 0; ci < 16; ++ci)
            #pragma unroll
            for (int k = 0; k < 5; ++k)
                v += xs[ci * 264 + p + k] * w1s[(co * 16 + ci) * 5 + k];
        h1s[co * 264 + p] = v > 0.f ? v : 0.f;
    }
    __syncthreads();

    int t = tBase + tid;
    if (t < 2167) {
        #pragma unroll
        for (int co = 0; co < 16; ++co) {
            float v = b2[co];
            #pragma unroll
            for (int ci = 0; ci < 16; ++ci)
                #pragma unroll
                for (int k = 0; k < 5; ++k)
                    v += h1s[ci * 264 + tid + k] * w2s[(co * 16 + ci) * 5 + k];
            v = v > 0.f ? v : 0.f;
            h2f[((size_t)n * 16 + co) * TP + t] = (f16)v;
        }
    }
}

// ---------------------------------------------------------------------------
// Kernel 2: conv3-as-GEMM + l1 + out, all WMMA. B tile double-buffered in LDS
// via async global->LDS (shared by all 8 waves); A streamed from global.
// Grid (16, 64), block 256 (8 waves x M=16 rows).
// ---------------------------------------------------------------------------
__global__ __launch_bounds__(256) void conv3_mlp_kernel(
    const f16* __restrict__ h2f, const f16* __restrict__ Bw,
    const f16* __restrict__ l1f, const f16* __restrict__ outf,
    const float* __restrict__ c3b, const float* __restrict__ l1b,
    const float* __restrict__ outb, float* __restrict__ y)
{
    __shared__ __attribute__((aligned(16))) f16 Bls[2 * 128 * BSTR]; // 20.0 KB
    __shared__ __attribute__((aligned(16))) f16 h3s[8 * 16 * 136];   // 34.8 KB
    __shared__ __attribute__((aligned(16))) f16 h4s[8 * 16 * 72];    // 18.4 KB

    const int tid   = threadIdx.x;
    const int lane  = tid & 31;
    const int wave  = tid >> 5;
    const int half  = lane >> 4;     // K-half selector in A/B operand layouts
    const int ln    = lane & 15;     // M-row (A) / N-col (B,C,D)
    const int n     = blockIdx.y;
    const int lBase = blockIdx.x * 128 + wave * 16;
    const f16* An   = h2f + (size_t)n * 16 * TP;

    // Per-thread slice of the cooperative B-tile stage: 32 bytes each.
    const int scol = tid >> 1;       // 0..127 output channel
    const int sj   = tid & 1;        // which 16-halve chunk of the 32-K slab

    // ---- conv3 GEMM: K = 64 steps of 32 (ci = ks>>2, k-chunk = (ks&3)*32) ----
    v8f acc[8] = {};
    {   // prologue: stage ks=0 into buffer 0
        const f16* g = Bw + (size_t)scol * 2048 + sj * 16;
        f16* l = Bls + scol * BSTR + sj * 16;
        copy16(g, l); copy16(g + 8, l + 8);
    }
    for (int ks = 0; ks < 64; ++ks) {
        wait_stage();                               // buf[ks&1] ready for all waves
        if (ks + 1 < 64) {                          // stage next tile (overlaps WMMA)
            const int ci1 = (ks + 1) >> 2;
            const int kk1 = ((ks + 1) & 3) * 32;
            const f16* g = Bw + (size_t)scol * 2048 + ci1 * 128 + kk1 + sj * 16;
            f16* l = Bls + ((ks + 1) & 1) * (128 * BSTR) + scol * BSTR + sj * 16;
            copy16(g, l); copy16(g + 8, l + 8);
        }
        const int ci = ks >> 2;
        const int kk = (ks & 3) * 32;
        const f16* ap = An + ci * TP + (lBase + ln) + kk + half * 8;
        __builtin_prefetch((const void*)(ap + TP), 0, 0);   // next-ci A stream
        v16h a = ld16h(ap, ap + 16);
        const f16* Bbuf = Bls + (ks & 1) * (128 * BSTR);
        // Preload all 8 B operand pairs first: ds_loads clause together and
        // the WMMA chain runs on partial dscnt waits instead of full stalls.
        v16h b[8];
        #pragma unroll
        for (int ct = 0; ct < 8; ++ct) {
            const f16* bp = Bbuf + (ct * 16 + ln) * BSTR + half * 8;
            b[ct] = ld16h(bp, bp + 16);
        }
        #pragma unroll
        for (int ct = 0; ct < 8; ++ct)
            acc[ct] = __builtin_amdgcn_wmma_f32_16x16x32_f16(
                false, a, false, b[ct], (short)0, acc[ct], false, false);
    }

    // bias + ReLU, transpose D (N=lanes, M=vgpr) -> row-major f16 tile in LDS
    f16* h3w = h3s + wave * 16 * 136;
    #pragma unroll
    for (int ct = 0; ct < 8; ++ct) {
        float bias = c3b[ct * 16 + ln];
        #pragma unroll
        for (int r = 0; r < 8; ++r) {
            float v = acc[ct][r] + bias;
            v = v > 0.f ? v : 0.f;
            h3w[(r + half * 8) * 136 + ct * 16 + ln] = (f16)v;
        }
    }
    __syncthreads();

    // ---- l1: (16 x 128) x (128 x 64), K = 4 steps ----
    v8f a2[4] = {};
    #pragma unroll
    for (int ks = 0; ks < 4; ++ks) {
        const f16* ap = h3w + ln * 136 + ks * 32 + half * 8;
        v16h a = ld16h(ap, ap + 16);
        v16h b[4];
        #pragma unroll
        for (int ct = 0; ct < 4; ++ct) {
            const f16* bp = l1f + (ct * 16 + ln) * 128 + ks * 32 + half * 8;
            b[ct] = ld16h(bp, bp + 16);
        }
        #pragma unroll
        for (int ct = 0; ct < 4; ++ct)
            a2[ct] = __builtin_amdgcn_wmma_f32_16x16x32_f16(
                false, a, false, b[ct], (short)0, a2[ct], false, false);
    }
    f16* h4w = h4s + wave * 16 * 72;
    #pragma unroll
    for (int ct = 0; ct < 4; ++ct) {
        float bias = l1b[ct * 16 + ln];
        #pragma unroll
        for (int r = 0; r < 8; ++r) {
            float v = a2[ct][r] + bias;
            v = v > 0.f ? v : 0.f;
            h4w[(r + half * 8) * 72 + ct * 16 + ln] = (f16)v;
        }
    }
    __syncthreads();

    // ---- out: (16 x 64) x (64 x 16pad), K = 2 steps ----
    v8f a3 = {};
    #pragma unroll
    for (int ks = 0; ks < 2; ++ks) {
        const f16* ap = h4w + ln * 72 + ks * 32 + half * 8;
        v16h a = ld16h(ap, ap + 16);
        const f16* bp = outf + ln * 64 + ks * 32 + half * 8;
        v16h b = ld16h(bp, bp + 16);
        a3 = __builtin_amdgcn_wmma_f32_16x16x32_f16(
            false, a, false, b, (short)0, a3, false, false);
    }
    if (ln < 5) {
        float bias = outb[ln];
        #pragma unroll
        for (int r = 0; r < 8; ++r) {
            int row = lBase + r + half * 8;
            y[((size_t)n * 2048 + row) * 5 + ln] = a3[r] + bias;
        }
    }
}

// ---------------------------------------------------------------------------
// Kernel 3: sequential Kalman filter; one thread per batch (64 threads).
// ---------------------------------------------------------------------------
__global__ void kalman_kernel(const float* __restrict__ y, float* __restrict__ out)
{
    int b = threadIdx.x;
    const float* yb = y + (size_t)b * 2048 * 5;
    float xp0 = yb[0], xp1 = yb[1];
    out[(size_t)b * 2048] = xp0;
    float s00 = 1.f, s01 = 0.f, s10 = 0.f, s11 = 1.f;
    for (int t = 1; t < 2048; ++t) {
        const float* yt = yb + t * 5;
        float z0 = yt[0], z1 = yt[1], q2 = yt[2], q3 = yt[3], q4 = yt[4];
        float a2 = q2 * q2, b4 = q4 * q4;
        float r00 = a2 * a2, r01 = a2 * q3, r11 = q3 * q3 + b4 * b4;
        float xm0 = fmaf(0.005f, xp1, xp0), xm1 = xp1;
        float sm00 = s00 + 0.005f * (s01 + s10) + 2.5e-5f * s11 + 0.1f;
        float sm01 = fmaf(0.005f, s11, s01);
        float sm10 = fmaf(0.005f, s11, s10);
        float sm11 = s11 + 0.1f;
        float S00 = sm00 + r00, S01 = sm01 + r01, S10 = sm10 + r01, S11 = sm11 + r11;
        float id = 1.f / (S00 * S11 - S01 * S10);
        float k00 = (sm00 * S11 - sm01 * S10) * id;
        float k01 = (sm01 * S00 - sm00 * S01) * id;
        float k10 = (sm10 * S11 - sm11 * S10) * id;
        float k11 = (sm11 * S00 - sm10 * S01) * id;
        float i0 = z0 - xm0, i1 = z1 - xm1;
        xp0 = xm0 + k00 * i0 + k01 * i1;
        xp1 = xm1 + k10 * i0 + k11 * i1;
        float n00 = (1.f - k00) * sm00 - k01 * sm10;
        float n01 = (1.f - k00) * sm01 - k01 * sm11;
        float n10 = (1.f - k11) * sm10 - k10 * sm00;
        float n11 = (1.f - k11) * sm11 - k10 * sm01;
        s00 = n00; s01 = n01; s10 = n10; s11 = n11;
        out[(size_t)b * 2048 + t] = xp0;
    }
}

// ---------------------------------------------------------------------------
extern "C" void kernel_launch(void* const* d_in, const int* in_sizes, int n_in,
                              void* d_out, int out_size, void* d_ws, size_t ws_size,
                              hipStream_t stream)
{
    (void)in_sizes; (void)n_in; (void)out_size; (void)ws_size;
    const float* x    = (const float*)d_in[0];
    const float* c1w  = (const float*)d_in[1];
    const float* c1b  = (const float*)d_in[2];
    const float* c2w  = (const float*)d_in[3];
    const float* c2b  = (const float*)d_in[4];
    const float* c3w  = (const float*)d_in[5];
    const float* c3b  = (const float*)d_in[6];
    const float* l1w  = (const float*)d_in[7];
    const float* l1b  = (const float*)d_in[8];
    const float* outw = (const float*)d_in[9];
    const float* outb = (const float*)d_in[10];
    float* out = (float*)d_out;

    char* ws = (char*)d_ws;
    size_t off = 0;
    auto take = [&](size_t bytes) {
        char* p = ws + off;
        off = (off + bytes + 255) & ~(size_t)255;
        return p;
    };
    f16*   h2f  = (f16*)take((size_t)64 * 16 * TP * 2);     // ~4.5 MB
    f16*   Bw   = (f16*)take((size_t)128 * 2048 * 2);       // 512 KB
    f16*   l1f  = (f16*)take((size_t)64 * 128 * 2);
    f16*   outf = (f16*)take((size_t)16 * 64 * 2);
    float* yb   = (float*)take((size_t)64 * 2048 * 5 * 4);  // ~2.6 MB

    prep_kernel<<<1024, 256, 0, stream>>>(c3w, l1w, outw, Bw, l1f, outf, h2f);
    conv12_kernel<<<dim3(9, 64), 256, 0, stream>>>(x, c1w, c1b, c2w, c2b, h2f);
    conv3_mlp_kernel<<<dim3(16, 64), 256, 0, stream>>>(h2f, Bw, l1f, outf,
                                                       c3b, l1b, outb, yb);
    kalman_kernel<<<1, 64, 0, stream>>>(yb, out);
}
